// LevitAttentionSubsample_6751688590164
// MI455X (gfx1250) — matrix operand order repaired and checked by reading, not compiled, not run on hardware
//
#include <hip/hip_runtime.h>

typedef __attribute__((ext_vector_type(16))) __bf16 v16bf;
typedef __attribute__((ext_vector_type(8)))  float  v8f;
typedef __attribute__((ext_vector_type(4)))  unsigned int v4u;
typedef __attribute__((ext_vector_type(8)))  int  v8i;
typedef __attribute__((ext_vector_type(4)))  int  v4i;

union FragB {
    v16bf v;
    uint4 q[2];
};

__device__ inline unsigned short f2bf(float f) {
    unsigned int u = __float_as_uint(f);
    unsigned int r = u + 0x7FFFu + ((u >> 16) & 1u);
    return (unsigned short)(r >> 16);
}

__device__ inline v8f wmma_bf16(const FragB& a, const FragB& b, v8f c) {
    return __builtin_amdgcn_wmma_f32_16x16x32_bf16(false, a.v, false, b.v,
                                                   (short)0, c, false, false);
}

#define SCALE_QK 0.25f

// ---------------------------------------------------------------------------
// Tensor Data Mover: 2D tile of bf16 rows (row_elems each) -> LDS with pad to
// 32-ushort row stride.  D# per CDNA5 ISA.
//   group0: count=1 | lds_addr | global_addr[56:0] | type=2
//   group1: data_size=2B, pad_enable, pad_interval=8dw, pad_amount=8dw,
//           tensor_dim0=row_elems, tensor_dim1=rows, tile=dims, stride0=row_elems
// 6-arg builtin form (clang-23 therock lane): (v4u, v8i, v4i, v4i, v8i, cpol)
// ---------------------------------------------------------------------------
__device__ inline void tdm_load_rows16(unsigned lds_off, const void* gptr, unsigned rows) {
    const unsigned row_elems = 16;
    unsigned long long ga = (unsigned long long)gptr;
    v4u g0;
    g0[0] = 1u;                                                  // count=1, user mode
    g0[1] = lds_off;                                             // lds_addr (bytes)
    g0[2] = (unsigned)(ga & 0xffffffffu);                        // global_addr lo
    g0[3] = (unsigned)((ga >> 32) & 0x01ffffffu) | 0x80000000u;  // ga hi | type=2
    v8i g1;
    g1[0] = (int)((1u << 16) | (1u << 20) | (2u << 22) | (7u << 25));
    //            data_size=2B  pad_en      pad_int=8dw  pad_amt=8dw
    g1[1] = (int)(row_elems << 16);           // tensor_dim0[15:0] in [31:16]
    g1[2] = (int)((rows & 0xffffu) << 16);    // dim0 hi=0, tensor_dim1[15:0]
    g1[3] = (int)(row_elems << 16);           // dim1 hi=0, tile_dim0
    g1[4] = (int)rows;                        // tile_dim1, tile_dim2=0
    g1[5] = (int)row_elems;                   // tensor_dim0_stride lo32
    g1[6] = 0;                                // stride0 hi, stride1 lo
    g1[7] = 0;                                // stride1 hi (unused, 2D tile)
    v4i z4 = {0, 0, 0, 0};
    v8i z8 = {0, 0, 0, 0, 0, 0, 0, 0};
    __builtin_amdgcn_tensor_load_to_lds(g0, g1, z4, z4, z8, 0);
}

// ---------------------------------------------------------------------------
// BN fold: s = gamma * rsqrt(var+eps), t = beta - mean*s
// ---------------------------------------------------------------------------
__global__ void bn_prep(const float* __restrict__ g, const float* __restrict__ be,
                        const float* __restrict__ me, const float* __restrict__ va,
                        float* __restrict__ s, float* __restrict__ t, int n) {
    int i = blockIdx.x * blockDim.x + threadIdx.x;
    if (i < n) {
        float sc = g[i] * rsqrtf(va[i] + 1e-5f);
        s[i] = sc;
        t[i] = be[i] - me[i] * sc;
    }
}

// ---------------------------------------------------------------------------
// Bias expand: [H,196,896] f32, cols >= 784 -> -1e30 (softmax-neutral pad)
// ---------------------------------------------------------------------------
__global__ void bias_expand(const float* __restrict__ biases, const int* __restrict__ idxs,
                            float* __restrict__ out, int n_off) {
    int i = blockIdx.x * blockDim.x + threadIdx.x;
    if (i >= 8 * 196 * 896) return;
    int h = i / (196 * 896);
    int rem = i % (196 * 896);
    int q = rem / 896, col = rem % 896;
    out[i] = (col < 784) ? biases[h * n_off + idxs[q * 784 + col]] : -1.0e30f;
}

// ---------------------------------------------------------------------------
// kv GEMM: X[100352,256] f32 x Wkv[640,256] f32 -> BN -> K bf16 [B,H,784,16],
// V bf16 transposed [B,H,64,784].  64x64 tile, 4 waves.
// ---------------------------------------------------------------------------
__global__ __launch_bounds__(128, 2)
void kv_gemm(const float* __restrict__ X, const float* __restrict__ W,
             const float* __restrict__ sK, const float* __restrict__ tK,
             unsigned short* __restrict__ Kg, unsigned short* __restrict__ Vg) {
    __shared__ unsigned short As[64 * 32];
    __shared__ unsigned short Bs[64 * 32];
    const int mtile = blockIdx.x, ntile = blockIdx.y;
    const int tid = threadIdx.x, wave = tid >> 5, lane = tid & 31;
    const int laneN = lane & 15;
    const int half8 = (lane < 16) ? 0 : 8;
    const int halfA = half8;                 // A-frag K offset (+8 for hi lanes)
    const int halfB = (lane < 16) ? 0 : 8;   // B-frag K offset (+8 for hi lanes)

    v8f zf = {};
    v8f acc[4];
#pragma unroll
    for (int i = 0; i < 4; i++) acc[i] = zf;

    for (int k0 = 0; k0 < 256; k0 += 32) {
        if (k0 + 32 < 256) {  // prefetch next A/B slabs -> global_prefetch_b8
            __builtin_prefetch(X + (size_t)(mtile * 64 + (tid >> 1)) * 256 + k0 + 32, 0, 1);
            __builtin_prefetch(W + (size_t)(ntile * 64 + (tid >> 1)) * 256 + k0 + 32, 0, 1);
        }
#pragma unroll
        for (int i = 0; i < 4; i++) {       // A tile: 64x32 f32 -> bf16
            int idx = tid + i * 128;
            int row = idx >> 3, c4 = idx & 7;
            const float4 f = ((const float4*)(X + (size_t)(mtile * 64 + row) * 256 + k0))[c4];
            uint2 u;
            u.x = (unsigned)f2bf(f.x) | ((unsigned)f2bf(f.y) << 16);
            u.y = (unsigned)f2bf(f.z) | ((unsigned)f2bf(f.w) << 16);
            *((uint2*)(As + row * 32 + c4 * 4)) = u;
        }
#pragma unroll
        for (int i = 0; i < 4; i++) {       // B tile (W rows are B columns)
            int idx = tid + i * 128;
            int row = idx >> 3, c4 = idx & 7;
            const float4 f = ((const float4*)(W + (size_t)(ntile * 64 + row) * 256 + k0))[c4];
            uint2 u;
            u.x = (unsigned)f2bf(f.x) | ((unsigned)f2bf(f.y) << 16);
            u.y = (unsigned)f2bf(f.z) | ((unsigned)f2bf(f.w) << 16);
            *((uint2*)(Bs + row * 32 + c4 * 4)) = u;
        }
        __syncthreads();
        FragB a;
        const uint4* ap = (const uint4*)(As + (wave * 16 + laneN) * 32);
        a.q[0] = ap[halfA >> 3];
        a.q[1] = ap[2 + (halfA >> 3)];
#pragma unroll
        for (int nt = 0; nt < 4; nt++) {
            FragB b;
            const uint4* bp = (const uint4*)(Bs + (nt * 16 + laneN) * 32);
            b.q[0] = bp[halfB >> 3];
            b.q[1] = bp[2 + (halfB >> 3)];
            acc[nt] = wmma_bf16(a, b, acc[nt]);
        }
        __syncthreads();
    }
#pragma unroll
    for (int nt = 0; nt < 4; nt++) {
        const int gn = ntile * 64 + nt * 16 + laneN;
        const float sc = sK[gn], sh = tK[gn];
        const int hd = gn / 80, offc = gn % 80;
#pragma unroll
        for (int r = 0; r < 8; r++) {
            const int gm = mtile * 64 + wave * 16 + r + half8;
            const int bb = gm / 784, row = gm % 784;
            const unsigned short val = f2bf(acc[nt][r] * sc + sh);
            if (offc < 16)
                Kg[(((size_t)bb * 8 + hd) * 784 + row) * 16 + offc] = val;
            else
                Vg[(((size_t)bb * 8 + hd) * 64 + (offc - 16)) * 784 + row] = val;
        }
    }
}

// ---------------------------------------------------------------------------
// q GEMM with ::2 spatial gather: out Q bf16 [B,H,196,16]
// ---------------------------------------------------------------------------
__global__ __launch_bounds__(128, 2)
void q_gemm(const float* __restrict__ X, const float* __restrict__ W,
            const float* __restrict__ sQ, const float* __restrict__ tQ,
            unsigned short* __restrict__ Qg) {
    __shared__ unsigned short As[64 * 32];
    __shared__ unsigned short Bs[64 * 32];
    const int mtile = blockIdx.x, ntile = blockIdx.y;
    const int tid = threadIdx.x, wave = tid >> 5, lane = tid & 31;
    const int laneN = lane & 15;
    const int half8 = (lane < 16) ? 0 : 8;
    const int halfA = half8;
    const int halfB = (lane < 16) ? 0 : 8;

    v8f zf = {};
    v8f acc[4];
#pragma unroll
    for (int i = 0; i < 4; i++) acc[i] = zf;

    for (int k0 = 0; k0 < 256; k0 += 32) {
#pragma unroll
        for (int i = 0; i < 4; i++) {
            int idx = tid + i * 128;
            int row = idx >> 3, c4 = idx & 7;
            int gm = mtile * 64 + row;
            int bb = gm / 196, n_ = gm % 196;
            int src = bb * 784 + (n_ / 14) * 56 + (n_ % 14) * 2;  // stride-2 subsample
            if (k0 + 32 < 256 && c4 == 0)
                __builtin_prefetch(X + (size_t)src * 256 + k0 + 32, 0, 1);
            const float4 f = ((const float4*)(X + (size_t)src * 256 + k0))[c4];
            uint2 u;
            u.x = (unsigned)f2bf(f.x) | ((unsigned)f2bf(f.y) << 16);
            u.y = (unsigned)f2bf(f.z) | ((unsigned)f2bf(f.w) << 16);
            *((uint2*)(As + row * 32 + c4 * 4)) = u;
        }
#pragma unroll
        for (int i = 0; i < 4; i++) {
            int idx = tid + i * 128;
            int row = idx >> 3, c4 = idx & 7;
            const float4 f = ((const float4*)(W + (size_t)(ntile * 64 + row) * 256 + k0))[c4];
            uint2 u;
            u.x = (unsigned)f2bf(f.x) | ((unsigned)f2bf(f.y) << 16);
            u.y = (unsigned)f2bf(f.z) | ((unsigned)f2bf(f.w) << 16);
            *((uint2*)(Bs + row * 32 + c4 * 4)) = u;
        }
        __syncthreads();
        FragB a;
        const uint4* ap = (const uint4*)(As + (wave * 16 + laneN) * 32);
        a.q[0] = ap[halfA >> 3];
        a.q[1] = ap[2 + (halfA >> 3)];
#pragma unroll
        for (int nt = 0; nt < 4; nt++) {
            FragB b;
            const uint4* bp = (const uint4*)(Bs + (nt * 16 + laneN) * 32);
            b.q[0] = bp[halfB >> 3];
            b.q[1] = bp[2 + (halfB >> 3)];
            acc[nt] = wmma_bf16(a, b, acc[nt]);
        }
        __syncthreads();
    }
#pragma unroll
    for (int nt = 0; nt < 4; nt++) {
        const int gn = ntile * 64 + nt * 16 + laneN;  // < 128
        const float sc = sQ[gn], sh = tQ[gn];
        const int hd = gn >> 4, offc = gn & 15;
#pragma unroll
        for (int r = 0; r < 8; r++) {
            const int gm = mtile * 64 + wave * 16 + r + half8;
            const int bb = gm / 196, n_ = gm % 196;
            Qg[(((size_t)bb * 8 + hd) * 196 + n_) * 16 + offc] = f2bf(acc[nt][r] * sc + sh);
        }
    }
}

// ---------------------------------------------------------------------------
// Fused attention per (b,h): K/V/Q resident in 213KB LDS (CDNA5 320KB/WGP).
// K and Q tiles staged by the Tensor Data Mover (pad feature builds the
// 32-ushort row stride); flash-style online softmax over 7 chunks of 128.
// ---------------------------------------------------------------------------
#define ATTN_SMEM 218112
__global__ __launch_bounds__(256, 1)
void attn_kernel(const unsigned short* __restrict__ Kg, const unsigned short* __restrict__ Vg,
                 const unsigned short* __restrict__ Qg, const float* __restrict__ BiasX,
                 unsigned short* __restrict__ Hbuf) {
    extern __shared__ char smem[];
    unsigned short* Kb = (unsigned short*)smem;             // [896][32]  (d padded to 32)
    unsigned short* Vt = (unsigned short*)(smem + 57344);   // [64][896]  (V transposed)
    unsigned short* Qb = (unsigned short*)(smem + 172032);  // [208][32]
    unsigned short* Pb = (unsigned short*)(smem + 185344);  // 8 x [16][128]

    const int bh = blockIdx.x;  // b*8 + h
    const int b = bh >> 3, h = bh & 7;
    const int tid = threadIdx.x;

    // zero all LDS (pads must be 0 for K/V/Q)
    {
        uint4 z4; z4.x = z4.y = z4.z = z4.w = 0;
        uint4* zp = (uint4*)smem;
        for (int i = tid; i < ATTN_SMEM / 16; i += 256) zp[i] = z4;
    }
    __syncthreads();

    // stage K and Q via TDM (wave 0 issues; TDM ignores EXEC so gate by wave)
    if (tid < 32) {
        tdm_load_rows16(0u,      Kg + (size_t)bh * 784 * 16, 784);
        tdm_load_rows16(172032u, Qg + (size_t)bh * 196 * 16, 196);
    }
    // stage Vt with regular loads (row length 392 dwords: not TDM-pad friendly)
    {
        const uint4* vr = (const uint4*)(Vg + (size_t)bh * 64 * 784);
        for (int i = tid; i < 64 * 98; i += 256) {  // 784 bf16 = 98 uint4 per row
            int d = i / 98, qo = i % 98;
            ((uint4*)(Vt + d * 896))[qo] = vr[d * 98 + qo];
        }
    }
    if (tid < 32) __builtin_amdgcn_s_wait_tensorcnt(0);
    __syncthreads();

    const int wave = tid >> 5, lane = tid & 31;
    const int laneN = lane & 15;
    const int half8 = (lane < 16) ? 0 : 8;
    const int halfA = half8;
    const int halfB = (lane < 16) ? 0 : 8;
    unsigned short* Pw = Pb + wave * 16 * 128;
    const float* bias0 = BiasX + (size_t)h * 196 * 896;
    v8f zf = {};

    for (int mb = wave; mb < 13; mb += 8) {   // 13 blocks of 16 q-rows (208 >= 196)
        const int q0 = mb * 16;
        FragB aq;
        {
            const uint4* p = (const uint4*)(Qb + (q0 + laneN) * 32);
            aq.q[0] = p[halfA >> 3];
            aq.q[1] = p[2 + (halfA >> 3)];
        }
        float mrun[8], lrun[8];
        v8f o[4];
#pragma unroll
        for (int i = 0; i < 4; i++) o[i] = zf;
#pragma unroll
        for (int r = 0; r < 8; r++) { mrun[r] = -3.0e38f; lrun[r] = 0.f; }

        for (int c = 0; c < 7; c++) {
            v8f s[8];
#pragma unroll
            for (int t = 0; t < 8; t++) {        // S tile = Q * K^T
                FragB bk;
                const uint4* p = (const uint4*)(Kb + (c * 128 + t * 16 + laneN) * 32);
                bk.q[0] = p[halfB >> 3];
                bk.q[1] = p[2 + (halfB >> 3)];
                s[t] = wmma_bf16(aq, bk, zf);
            }
            // scale + bias
#pragma unroll
            for (int r = 0; r < 8; r++) {
                const int q = q0 + r + half8;
#pragma unroll
                for (int t = 0; t < 8; t++) {
                    float bia = (q < 196) ? bias0[(size_t)q * 896 + c * 128 + t * 16 + laneN]
                                          : -1.0e30f;
                    s[t][r] = s[t][r] * SCALE_QK + bia;
                }
            }
            // row max (across 8 tiles, then across 16 lanes of each half)
            float cm[8];
#pragma unroll
            for (int r = 0; r < 8; r++) {
                float m = s[0][r];
#pragma unroll
                for (int t = 1; t < 8; t++) m = fmaxf(m, s[t][r]);
                cm[r] = m;
            }
#pragma unroll
            for (int msk = 1; msk <= 8; msk <<= 1)
#pragma unroll
                for (int r = 0; r < 8; r++) cm[r] = fmaxf(cm[r], __shfl_xor(cm[r], msk, 32));

            float mnew[8], scl[8], cs[8];
#pragma unroll
            for (int r = 0; r < 8; r++) {
                mnew[r] = fmaxf(mrun[r], cm[r]);
                scl[r] = __expf(mrun[r] - mnew[r]);
                cs[r] = 0.f;
            }
#pragma unroll
            for (int r = 0; r < 8; r++)
#pragma unroll
                for (int t = 0; t < 8; t++) {
                    float p = __expf(s[t][r] - mnew[r]);
                    s[t][r] = p;
                    cs[r] += p;
                }
#pragma unroll
            for (int msk = 1; msk <= 8; msk <<= 1)
#pragma unroll
                for (int r = 0; r < 8; r++) cs[r] += __shfl_xor(cs[r], msk, 32);
#pragma unroll
            for (int r = 0; r < 8; r++) {
                lrun[r] = lrun[r] * scl[r] + cs[r];
                mrun[r] = mnew[r];
            }
#pragma unroll
            for (int nt = 0; nt < 4; nt++)
#pragma unroll
                for (int r = 0; r < 8; r++) o[nt][r] *= scl[r];

            // restage P (D-layout -> A-layout) through per-wave LDS
#pragma unroll
            for (int r = 0; r < 8; r++) {
                const int row = r + half8;
#pragma unroll
                for (int t = 0; t < 8; t++) Pw[row * 128 + t * 16 + laneN] = f2bf(s[t][r]);
            }
            // O += P * V   (same-wave LDS ops are in-order; no block barrier needed)
#pragma unroll
            for (int kb = 0; kb < 4; kb++) {
                FragB ap;
                const uint4* pp = (const uint4*)(Pw + laneN * 128 + kb * 32);
                ap.q[0] = pp[halfA >> 3];
                ap.q[1] = pp[2 + (halfA >> 3)];
#pragma unroll
                for (int nt = 0; nt < 4; nt++) {
                    FragB bv;
                    const uint4* vp =
                        (const uint4*)(Vt + (nt * 16 + laneN) * 896 + c * 128 + kb * 32);
                    bv.q[0] = vp[halfB >> 3];
                    bv.q[1] = vp[2 + (halfB >> 3)];
                    o[nt] = wmma_bf16(ap, bv, o[nt]);
                }
            }
        }
        // epilogue: normalize, hardswish, store bf16 to [B,196,512]
#pragma unroll
        for (int nt = 0; nt < 4; nt++)
#pragma unroll
            for (int r = 0; r < 8; r++) {
                const int q = q0 + r + half8;
                if (q < 196) {
                    float v = o[nt][r] / lrun[r];
                    float hs = v * fminf(fmaxf(v + 3.f, 0.f), 6.f) * (1.f / 6.f);
                    Hbuf[((size_t)b * 196 + q) * 512 + h * 64 + nt * 16 + laneN] = f2bf(hs);
                }
            }
    }
}

// ---------------------------------------------------------------------------
// proj GEMM: Hbuf bf16 [25088,512] x Wproj[384,512] f32 -> BN -> f32 out
// ---------------------------------------------------------------------------
__global__ __launch_bounds__(128, 2)
void proj_gemm(const unsigned short* __restrict__ A, const float* __restrict__ W,
               const float* __restrict__ sP, const float* __restrict__ tP,
               float* __restrict__ out) {
    __shared__ unsigned short As[64 * 32];
    __shared__ unsigned short Bs[64 * 32];
    const int mtile = blockIdx.x, ntile = blockIdx.y;
    const int tid = threadIdx.x, wave = tid >> 5, lane = tid & 31;
    const int laneN = lane & 15;
    const int half8 = (lane < 16) ? 0 : 8;
    const int halfA = half8;
    const int halfB = (lane < 16) ? 0 : 8;

    v8f zf = {};
    v8f acc[4];
#pragma unroll
    for (int i = 0; i < 4; i++) acc[i] = zf;

    for (int k0 = 0; k0 < 512; k0 += 32) {
        if (k0 + 32 < 512) {
            __builtin_prefetch(A + (size_t)(mtile * 64 + (tid >> 1)) * 512 + k0 + 32, 0, 1);
            __builtin_prefetch(W + (size_t)(ntile * 64 + (tid >> 1)) * 512 + k0 + 32, 0, 1);
        }
#pragma unroll
        for (int i = 0; i < 2; i++) {  // A already bf16: straight 16B copies
            int idx = tid + i * 128;   // 0..255 uint4
            int row = idx >> 2, c8 = idx & 3;
            ((uint4*)(As + row * 32))[c8] =
                ((const uint4*)(A + (size_t)(mtile * 64 + row) * 512 + k0))[c8];
        }
#pragma unroll
        for (int i = 0; i < 4; i++) {
            int idx = tid + i * 128;
            int row = idx >> 3, c4 = idx & 7;
            const float4 f = ((const float4*)(W + (size_t)(ntile * 64 + row) * 512 + k0))[c4];
            uint2 u;
            u.x = (unsigned)f2bf(f.x) | ((unsigned)f2bf(f.y) << 16);
            u.y = (unsigned)f2bf(f.z) | ((unsigned)f2bf(f.w) << 16);
            *((uint2*)(Bs + row * 32 + c4 * 4)) = u;
        }
        __syncthreads();
        FragB a;
        const uint4* ap = (const uint4*)(As + (wave * 16 + laneN) * 32);
        a.q[0] = ap[halfA >> 3];
        a.q[1] = ap[2 + (halfA >> 3)];
#pragma unroll
        for (int nt = 0; nt < 4; nt++) {
            FragB b;
            const uint4* bp = (const uint4*)(Bs + (nt * 16 + laneN) * 32);
            b.q[0] = bp[halfB >> 3];
            b.q[1] = bp[2 + (halfB >> 3)];
            acc[nt] = wmma_bf16(a, b, acc[nt]);
        }
        __syncthreads();
    }
#pragma unroll
    for (int nt = 0; nt < 4; nt++) {
        const int gn = ntile * 64 + nt * 16 + laneN;  // < 384
        const float sc = sP[gn], sh = tP[gn];
#pragma unroll
        for (int r = 0; r < 8; r++) {
            const int gm = mtile * 64 + wave * 16 + r + half8;
            out[(size_t)gm * 384 + gn] = acc[nt][r] * sc + sh;
        }
    }
}

// ---------------------------------------------------------------------------
extern "C" void kernel_launch(void* const* d_in, const int* in_sizes, int n_in,
                              void* d_out, int out_size, void* d_ws, size_t ws_size,
                              hipStream_t stream) {
    const float* hidden   = (const float*)d_in[0];
    const float* w_kv     = (const float*)d_in[1];
    const float* kv_gamma = (const float*)d_in[2];
    const float* kv_beta  = (const float*)d_in[3];
    const float* kv_mean  = (const float*)d_in[4];
    const float* kv_var   = (const float*)d_in[5];
    const float* w_q      = (const float*)d_in[6];
    const float* q_gamma  = (const float*)d_in[7];
    const float* q_beta   = (const float*)d_in[8];
    const float* q_mean   = (const float*)d_in[9];
    const float* q_var    = (const float*)d_in[10];
    const float* w_proj   = (const float*)d_in[11];
    const float* p_gamma  = (const float*)d_in[12];
    const float* p_beta   = (const float*)d_in[13];
    const float* p_mean   = (const float*)d_in[14];
    const float* p_var    = (const float*)d_in[15];
    const float* biases   = (const float*)d_in[16];
    const int*   idxs     = (const int*)d_in[17];
    float* out = (float*)d_out;
    const int n_off = in_sizes[16] / 8;
    (void)n_in; (void)out_size; (void)ws_size;

    char* ws = (char*)d_ws;
    size_t off = 0;
    auto alloc = [&](size_t bytes) -> void* {
        void* p = ws + off;
        off += (bytes + 255) & ~(size_t)255;
        return p;
    };
    float* sKv = (float*)alloc(640 * 4);
    float* tKv = (float*)alloc(640 * 4);
    float* sQ  = (float*)alloc(128 * 4);
    float* tQ  = (float*)alloc(128 * 4);
    float* sP  = (float*)alloc(384 * 4);
    float* tP  = (float*)alloc(384 * 4);
    unsigned short* Kg = (unsigned short*)alloc((size_t)128 * 8 * 784 * 16 * 2);
    unsigned short* Vg = (unsigned short*)alloc((size_t)128 * 8 * 64 * 784 * 2);
    unsigned short* Qg = (unsigned short*)alloc((size_t)128 * 8 * 196 * 16 * 2);
    float* BiasX = (float*)alloc((size_t)8 * 196 * 896 * 4);
    unsigned short* Hb = (unsigned short*)alloc((size_t)128 * 196 * 512 * 2);

    bn_prep<<<dim3(3), dim3(256), 0, stream>>>(kv_gamma, kv_beta, kv_mean, kv_var, sKv, tKv, 640);
    bn_prep<<<dim3(1), dim3(256), 0, stream>>>(q_gamma, q_beta, q_mean, q_var, sQ, tQ, 128);
    bn_prep<<<dim3(2), dim3(256), 0, stream>>>(p_gamma, p_beta, p_mean, p_var, sP, tP, 384);
    bias_expand<<<dim3((8 * 196 * 896 + 255) / 256), dim3(256), 0, stream>>>(biases, idxs, BiasX, n_off);

    kv_gemm<<<dim3(1568, 10), dim3(128), 0, stream>>>(hidden, w_kv, sKv, tKv, Kg, Vg);
    q_gemm<<<dim3(392, 2), dim3(128), 0, stream>>>(hidden, w_q, sQ, tQ, Qg);

    (void)hipFuncSetAttribute(reinterpret_cast<const void*>(attn_kernel),
                              hipFuncAttributeMaxDynamicSharedMemorySize, ATTN_SMEM);
    attn_kernel<<<dim3(1024), dim3(256), ATTN_SMEM, stream>>>(Kg, Vg, Qg, BiasX, Hb);

    proj_gemm<<<dim3(392, 6), dim3(128), 0, stream>>>(Hb, w_proj, sP, tP, out);
}
